// Decoder_10934986736069
// MI455X (gfx1250) — matrix operand order repaired and checked by reading, not compiled, not run on hardware
//
#include <hip/hip_runtime.h>
#include <hip/hip_bf16.h>

// ---------------------------------------------------------------------------
// LAS attention decoder for MI455X (gfx1250, wave32, WMMA).
// All dense GEMMs run through v_wmma_f32_16x16x32_bf16 with one wave per
// 16x16 output tile. Weights are pre-converted once per call into bf16
// "W-transposed" (N x K row-major) layout. The B (weight) operand is staged
// through LDS with double-buffered GLOBAL_LOAD_ASYNC_TO_LDS_B128 (ASYNCcnt)
// when the toolchain exposes the builtins; otherwise direct b128 loads.
// Activations stay f32 and are converted to bf16 in-register for the A
// fragment. Attention (batched mat-vecs + softmax over T=512) stays on VALU
// with one workgroup per batch row. Working set is L2-resident (192MB L2).
// ---------------------------------------------------------------------------

typedef __bf16 bf16_t;
typedef __attribute__((ext_vector_type(16))) __bf16 v16bf;
typedef __attribute__((ext_vector_type(8)))  float  v8f;
typedef __attribute__((ext_vector_type(4)))  int    v4i;

#if __has_builtin(__builtin_amdgcn_global_load_async_to_lds_b128) && \
    __has_builtin(__builtin_amdgcn_s_wait_asynccnt)
#define HAVE_ASYNC_LDS 1
// address-space-qualified int4 pointers matching the builtin's signature
typedef v4i __attribute__((address_space(1))) *gptr4_t;
typedef v4i __attribute__((address_space(3))) *lptr4_t;
#else
#define HAVE_ASYNC_LDS 0
#endif

#define NB      32
#define TT      512
#define LLEN    200
#define VOCABSZ 10000
#define EMBD    256
#define KEYD    256
#define DECH    512
#define ENCOUT  1024

#define GEMM_WAVES 8            // waves per block (wave32 -> 256 threads)

// ---------------------------------------------------------------------------
// Generic WMMA GEMM:  C(MxN) = act( A(MxK,f32,lda) * WT(NxK,bf16)^T + bias )
// One wave computes one 16x16 tile, looping K in chunks of 32.
// A fragment (16x32 bf16): lane l: m = m0+(l&15), half = l>>4;
//   elements 0..7  <- A[m, kc + half*16 + e]
//   elements 8..15 <- A[m, kc + half*16 + 8 + e]   (contiguous, matches B)
// B fragment (32x16 bf16): lane l: n = n0+(l&15);
//   element e      <- WT[n, kc + half*16 + e]      (contiguous in WT row)
// C/D f32: VGPR r -> row m0 + r + 8*half, col n0 + (l&15).
// ---------------------------------------------------------------------------
__global__ void gemm_wmma_bf16(const float* __restrict__ A, long long lda,
                               const bf16_t* __restrict__ WT,
                               const float* __restrict__ bias,
                               float* __restrict__ Cf, bf16_t* __restrict__ Cb,
                               long long ldc, int M, int N, int K, int act)
{
#if HAVE_ASYNC_LDS
    // per-wave double buffer: 2 chunks x (16 rows x 64B) = 2KB
    __shared__ __align__(64) char btile[GEMM_WAVES * 2048];
#endif
    const int tiles_n = N >> 4;
    const int total   = (M >> 4) * tiles_n;
    const int wave    = blockIdx.x * (blockDim.x >> 5) + (threadIdx.x >> 5);
    if (wave >= total) return;                  // wave-uniform: EXEC stays all-1
    const int m0   = (wave / tiles_n) << 4;
    const int n0   = (wave % tiles_n) << 4;
    const int lane = threadIdx.x & 31;
    const int hh   = lane >> 4;
    const int lm   = lane & 15;

    const float*  Arow = A  + (size_t)(m0 + lm) * (size_t)lda;
    const bf16_t* Brow = WT + (size_t)(n0 + lm) * (size_t)K + hh * 16;

    v8f acc = {0.f, 0.f, 0.f, 0.f, 0.f, 0.f, 0.f, 0.f};

#if HAVE_ASYNC_LDS
    char* mybuf = btile + (threadIdx.x >> 5) * 2048;
    // async-load roles: lane covers row lm, 32B segment hh of the 64B row
    const char* gRow = (const char*)(WT + (size_t)(n0 + lm) * (size_t)K) + hh * 32;
    auto issue = [&](int kc, int buf) {
        const char* g = gRow + (size_t)kc * sizeof(bf16_t);
        char*       l = mybuf + buf * 1024 + lm * 64 + hh * 32;
        __builtin_amdgcn_global_load_async_to_lds_b128(
            (gptr4_t)(g),      (lptr4_t)(l),      0, 0);
        __builtin_amdgcn_global_load_async_to_lds_b128(
            (gptr4_t)(g + 16), (lptr4_t)(l + 16), 0, 0);
    };
    issue(0, 0);
    int buf = 0;
    for (int kc = 0; kc < K; kc += 32) {
        const int kcn = (kc + 32 < K) ? kc + 32 : kc;   // clamped prefetch issue
        issue(kcn, buf ^ 1);

        const float* ap0 = Arow + kc + hh * 16;
        v16bf af;
        #pragma unroll
        for (int i = 0; i < 8; ++i) {
            af[i]     = (bf16_t)ap0[i];
            af[8 + i] = (bf16_t)ap0[8 + i];
        }
        __builtin_amdgcn_s_wait_asynccnt(2);   // current chunk's 2 ops retired
        v16bf bfv = *(const v16bf*)(mybuf + buf * 1024 + lm * 64 + hh * 32);
        acc = __builtin_amdgcn_wmma_f32_16x16x32_bf16(
                  false, af, false, bfv, (short)0, acc, false, false);
        buf ^= 1;
    }
#else
    for (int kc = 0; kc < K; kc += 32) {
        const float* ap0 = Arow + kc + hh * 16;
        v16bf af;
        #pragma unroll
        for (int i = 0; i < 8; ++i) {
            af[i]     = (bf16_t)ap0[i];
            af[8 + i] = (bf16_t)ap0[8 + i];
        }
        v16bf bfv = *(const v16bf*)(Brow + kc);   // 32B -> two b128 loads
        __builtin_prefetch(Brow + kc + 32, 0, 3); // WGP-scope prefetch
        acc = __builtin_amdgcn_wmma_f32_16x16x32_bf16(
                  false, af, false, bfv, (short)0, acc, false, false);
    }
#endif

    const int   col = n0 + lm;
    const float bb  = bias ? bias[col] : 0.f;
    #pragma unroll
    for (int r = 0; r < 8; ++r) {
        const int row = m0 + r + hh * 8;
        float v = acc[r] + bb;
        if (act == 1) v = fminf(1.f, fmaxf(-1.f, v));   // Hardtanh
        const size_t idx = (size_t)row * (size_t)ldc + (size_t)col;
        if (Cb) Cb[idx] = (bf16_t)v;
        else    Cf[idx] = v;
    }
}

// ---------------------------------------------------------------------------
// Attention: one workgroup (256 threads) per batch row n.
// energy[t] = k[n,t,:] . q[n,:], masked by lens; softmax over T=512;
// ctx[d] = sum_t att[t] * v[n,t,d].  Optionally records att for n==0.
// ---------------------------------------------------------------------------
__global__ void attend_kernel(const bf16_t* __restrict__ kmat,
                              const bf16_t* __restrict__ vmat,
                              const float* __restrict__ q,
                              const int* __restrict__ lens,
                              float* __restrict__ ctx_dst, int ctx_stride,
                              float* __restrict__ att_out)
{
    __shared__ float att[TT];
    __shared__ float qs[KEYD];
    __shared__ float red[256];

    const int n   = blockIdx.x;
    const int tid = threadIdx.x;

    qs[tid] = q[n * KEYD + tid];
    __syncthreads();

    const int len = lens[n];
    const bf16_t* kb = kmat + (size_t)n * TT * KEYD;

    float lmax = -1e30f;
    for (int t = tid; t < TT; t += 256) {
        const bf16_t* kr = kb + (size_t)t * KEYD;
        float e = 0.f;
        #pragma unroll 8
        for (int d = 0; d < KEYD; ++d) e += (float)kr[d] * qs[d];
        e = (t < len) ? e : -1e9f;
        att[t] = e;
        lmax = fmaxf(lmax, e);
    }
    red[tid] = lmax;
    __syncthreads();
    for (int s = 128; s > 0; s >>= 1) {
        if (tid < s) red[tid] = fmaxf(red[tid], red[tid + s]);
        __syncthreads();
    }
    const float mx = red[0];
    __syncthreads();

    float lsum = 0.f;
    for (int t = tid; t < TT; t += 256) {
        float e = __expf(att[t] - mx);
        att[t] = e;
        lsum += e;
    }
    red[tid] = lsum;
    __syncthreads();
    for (int s = 128; s > 0; s >>= 1) {
        if (tid < s) red[tid] += red[tid + s];
        __syncthreads();
    }
    const float inv = 1.f / red[0];
    __syncthreads();
    for (int t = tid; t < TT; t += 256) att[t] *= inv;
    __syncthreads();

    // ctx: thread tid owns dim d = tid (KEYD == 256 == blockDim)
    const bf16_t* vb = vmat + (size_t)n * TT * KEYD;
    float acc = 0.f;
    #pragma unroll 4
    for (int t = 0; t < TT; ++t)
        acc += att[t] * (float)vb[(size_t)t * KEYD + tid];
    ctx_dst[(size_t)n * ctx_stride + tid] = acc;

    if (att_out && n == 0)
        for (int t = tid; t < TT; t += 256) att_out[t] = att[t];
}

// ---------------------------------------------------------------------------
// LSTM gate nonlinearity (split i,f,g,o), c update, h written to two places.
// ---------------------------------------------------------------------------
__device__ __forceinline__ float sigmoidf_(float x) { return 1.f / (1.f + __expf(-x)); }

__global__ void lstm_apply(const float* __restrict__ gates, float* __restrict__ c,
                           float* __restrict__ h0, int ld0,
                           float* __restrict__ h1, int ld1, int H)
{
    const int idx = blockIdx.x * blockDim.x + threadIdx.x;
    if (idx >= NB * H) return;
    const int n = idx / H, j = idx - n * H;
    const float* g = gates + (size_t)n * 4 * H;
    const float iv = sigmoidf_(g[j]);
    const float fv = sigmoidf_(g[H + j]);
    const float gv = tanhf(g[2 * H + j]);
    const float ov = sigmoidf_(g[3 * H + j]);
    const float cn = fv * c[idx] + iv * gv;
    c[idx] = cn;
    const float hv = ov * tanhf(cn);
    h0[(size_t)n * ld0 + j] = hv;
    h1[(size_t)n * ld1 + j] = hv;
}

// ---------------------------------------------------------------------------
// Prep kernels (run every call; deterministic).
// ---------------------------------------------------------------------------
__global__ void transpose_cvt(const float* __restrict__ src, bf16_t* __restrict__ dst,
                              int I, int O)   // src (I x O) -> dst (O x I) bf16
{
    const int idx = blockIdx.x * blockDim.x + threadIdx.x;
    if (idx >= I * O) return;
    const int o = idx / I, i = idx - o * I;
    dst[idx] = (bf16_t)src[(size_t)i * O + o];
}

__global__ void concat2_cvt(const float* __restrict__ A, const float* __restrict__ B,
                            bf16_t* __restrict__ dst, int J, int K1, int K2)
{
    const int KT  = K1 + K2;
    const int idx = blockIdx.x * blockDim.x + threadIdx.x;
    if (idx >= J * KT) return;
    const int j = idx / KT, k = idx - j * KT;
    dst[idx] = (bf16_t)((k < K1) ? A[(size_t)j * K1 + k]
                                 : B[(size_t)j * K2 + (k - K1)]);
}

__global__ void cvt_bf16(const float* __restrict__ src, bf16_t* __restrict__ dst, int n)
{
    const int idx = blockIdx.x * blockDim.x + threadIdx.x;
    if (idx < n) dst[idx] = (bf16_t)src[idx];
}

__global__ void add2_f32(const float* __restrict__ a, const float* __restrict__ b,
                         float* __restrict__ dst, int n)
{
    const int idx = blockIdx.x * blockDim.x + threadIdx.x;
    if (idx < n) dst[idx] = a[idx] + b[idx];
}

__global__ void zero_f32(float* __restrict__ p, int n)
{
    const int idx = blockIdx.x * blockDim.x + threadIdx.x;
    if (idx < n) p[idx] = 0.f;
}

__global__ void embed_gather(const float* __restrict__ E, const int* __restrict__ text,
                             float* __restrict__ xh1, int s)
{
    const int idx = blockIdx.x * blockDim.x + threadIdx.x;
    if (idx >= NB * EMBD) return;
    const int n = idx >> 8, d = idx & 255;
    const int tok = (s == 0) ? 0 : text[n * LLEN + (s - 1)];
    xh1[(size_t)n * (EMBD + KEYD + DECH) + d] = E[(size_t)tok * EMBD + d];
}

// ---------------------------------------------------------------------------
extern "C" void kernel_launch(void* const* d_in, const int* in_sizes, int n_in,
                              void* d_out, int out_size, void* d_ws, size_t ws_size,
                              hipStream_t stream)
{
    (void)in_sizes; (void)n_in; (void)out_size; (void)ws_size;

    const float* enc   = (const float*)d_in[0];
    const int*   text  = (const int*)  d_in[1];
    const int*   lens  = (const int*)  d_in[2];
    const float* E     = (const float*)d_in[3];
    const float* Wk    = (const float*)d_in[4];
    const float* bk    = (const float*)d_in[5];
    const float* Wv    = (const float*)d_in[6];
    const float* bv    = (const float*)d_in[7];
    const float* Wq    = (const float*)d_in[8];
    const float* bq    = (const float*)d_in[9];
    const float* Wih1  = (const float*)d_in[10];
    const float* Whh1  = (const float*)d_in[11];
    const float* bih1  = (const float*)d_in[12];
    const float* bhh1  = (const float*)d_in[13];
    const float* Wih2  = (const float*)d_in[14];
    const float* Whh2  = (const float*)d_in[15];
    const float* bih2  = (const float*)d_in[16];
    const float* bhh2  = (const float*)d_in[17];
    const float* Wfc   = (const float*)d_in[18];
    const float* bfc   = (const float*)d_in[19];
    const float* bchar = (const float*)d_in[20];
    float* out = (float*)d_out;

    // ---- workspace bump allocator (256B aligned) ----
    char*  base = (char*)d_ws;
    size_t off  = 0;
    auto alloc = [&](size_t bytes) -> void* {
        void* p = base + off;
        off += (bytes + 255) & ~(size_t)255;
        return p;
    };
    bf16_t* WkT  = (bf16_t*)alloc((size_t)KEYD * ENCOUT * sizeof(bf16_t));
    bf16_t* WvT  = (bf16_t*)alloc((size_t)KEYD * ENCOUT * sizeof(bf16_t));
    bf16_t* WqT  = (bf16_t*)alloc((size_t)KEYD * EMBD   * sizeof(bf16_t));
    bf16_t* WfcT = (bf16_t*)alloc((size_t)EMBD * 2 * KEYD * sizeof(bf16_t));
    bf16_t* Wc1  = (bf16_t*)alloc((size_t)4 * DECH * (EMBD + KEYD + DECH) * sizeof(bf16_t));
    bf16_t* Wc2  = (bf16_t*)alloc((size_t)4 * KEYD * (DECH + KEYD) * sizeof(bf16_t));
    bf16_t* Ebf  = (bf16_t*)alloc((size_t)VOCABSZ * EMBD * sizeof(bf16_t));
    bf16_t* kbf  = (bf16_t*)alloc((size_t)NB * TT * KEYD * sizeof(bf16_t));
    bf16_t* vbf  = (bf16_t*)alloc((size_t)NB * TT * KEYD * sizeof(bf16_t));
    float*  b1   = (float*)alloc(4 * DECH * sizeof(float));
    float*  b2   = (float*)alloc(4 * KEYD * sizeof(float));
    const int stateN = NB * (1024 + 768 + 512 + 512 + 256 + 256 + 2048 + 1024 + 256);
    float* stateF = (float*)alloc((size_t)stateN * sizeof(float));
    float* xh1  = stateF;             // 32 x 1024 : [emb | ctx1 | h1]
    float* xh2  = xh1 + NB * 1024;    // 32 x 768  : [h1 | h2]
    float* xfc  = xh2 + NB * 768;     // 32 x 512  : [h2 | ctx2]
    float* c1   = xfc + NB * 512;     // 32 x 512
    float* c2   = c1  + NB * 512;     // 32 x 256
    float* qbuf = c2  + NB * 256;     // 32 x 256
    float* g1   = qbuf + NB * 256;    // 32 x 2048
    float* g2   = g1  + NB * 2048;    // 32 x 1024
    float* fco  = g2  + NB * 1024;    // 32 x 256

    auto gemm = [&](const float* A, long long lda, const bf16_t* WT, const float* bias,
                    float* Cf, bf16_t* Cb, long long ldc, int M, int N, int K, int act) {
        const int tiles  = (M / 16) * (N / 16);
        const int blocks = (tiles + GEMM_WAVES - 1) / GEMM_WAVES;
        gemm_wmma_bf16<<<blocks, GEMM_WAVES * 32, 0, stream>>>(A, lda, WT, bias, Cf, Cb, ldc, M, N, K, act);
    };
    auto g1d = [](int n) { return (n + 255) / 256; };

    // ---- per-call prep: weights -> bf16 WT-form, bias sums, zero state ----
    transpose_cvt<<<g1d(ENCOUT * KEYD), 256, 0, stream>>>(Wk,  WkT,  ENCOUT, KEYD);
    transpose_cvt<<<g1d(ENCOUT * KEYD), 256, 0, stream>>>(Wv,  WvT,  ENCOUT, KEYD);
    transpose_cvt<<<g1d(EMBD * KEYD),   256, 0, stream>>>(Wq,  WqT,  EMBD,   KEYD);
    transpose_cvt<<<g1d(2 * KEYD * EMBD), 256, 0, stream>>>(Wfc, WfcT, 2 * KEYD, EMBD);
    concat2_cvt<<<g1d(4 * DECH * 1024), 256, 0, stream>>>(Wih1, Whh1, Wc1, 4 * DECH, EMBD + KEYD, DECH);
    concat2_cvt<<<g1d(4 * KEYD * 768),  256, 0, stream>>>(Wih2, Whh2, Wc2, 4 * KEYD, DECH, KEYD);
    cvt_bf16<<<g1d(VOCABSZ * EMBD), 256, 0, stream>>>(E, Ebf, VOCABSZ * EMBD);
    add2_f32<<<g1d(4 * DECH), 256, 0, stream>>>(bih1, bhh1, b1, 4 * DECH);
    add2_f32<<<g1d(4 * KEYD), 256, 0, stream>>>(bih2, bhh2, b2, 4 * KEYD);
    zero_f32<<<g1d(stateN), 256, 0, stream>>>(stateF, stateN);

    // ---- KV precompute: (16384 x 1024) x (1024 x 256) via WMMA -> bf16 ----
    gemm(enc, ENCOUT, WkT, bk, nullptr, kbf, KEYD, NB * TT, KEYD, ENCOUT, 0);
    gemm(enc, ENCOUT, WvT, bv, nullptr, vbf, KEYD, NB * TT, KEYD, ENCOUT, 0);

    // ---- 200 sequential decode steps ----
    float* attns_base = out + (size_t)NB * LLEN * VOCABSZ;   // (L, T) tail
    for (int s = 0; s < LLEN; ++s) {
        embed_gather<<<g1d(NB * EMBD), 256, 0, stream>>>(E, text, xh1, s);
        // q1 = emb @ Wq + bq
        gemm(xh1, 1024, WqT, bq, qbuf, nullptr, KEYD, NB, KEYD, EMBD, 0);
        // ctx1 -> xh1[:, 256:512]
        attend_kernel<<<NB, 256, 0, stream>>>(kbf, vbf, qbuf, lens, xh1 + 256, 1024, nullptr);
        // LSTM1: gates = [emb|ctx1|h1] @ [Wih1|Whh1]^T + (b_ih1+b_hh1)
        gemm(xh1, 1024, Wc1, b1, g1, nullptr, 4 * DECH, NB, 4 * DECH, 1024, 0);
        lstm_apply<<<g1d(NB * DECH), 256, 0, stream>>>(g1, c1, xh1 + 512, 1024, xh2, 768, DECH);
        // LSTM2: gates = [h1|h2] @ [Wih2|Whh2]^T + (b_ih2+b_hh2)
        gemm(xh2, 768, Wc2, b2, g2, nullptr, 4 * KEYD, NB, 4 * KEYD, 768, 0);
        lstm_apply<<<g1d(NB * KEYD), 256, 0, stream>>>(g2, c2, xh2 + 512, 768, xfc, 512, KEYD);
        // q2 = h2 @ Wq + bq
        gemm(xh2 + 512, 768, WqT, bq, qbuf, nullptr, KEYD, NB, KEYD, KEYD, 0);
        // ctx2 -> xfc[:, 256:512]; record att for n==0
        attend_kernel<<<NB, 256, 0, stream>>>(kbf, vbf, qbuf, lens, xfc + 256, 512,
                                              attns_base + (size_t)s * TT);
        // fco = hardtanh([h2|ctx2] @ Wfc + bfc)
        gemm(xfc, 512, WfcT, bfc, fco, nullptr, EMBD, NB, EMBD, 2 * KEYD, 1);
        // pred = fco @ E^T + b_char  -> out[n, s, :]  (ldc = L*VOCAB)
        gemm(fco, EMBD, Ebf, bchar, out + (size_t)s * VOCABSZ, nullptr,
             (long long)LLEN * VOCABSZ, NB, VOCABSZ, EMBD, 0);
    }
}
